// _MockTurboCoreAttention_50964081935268
// MI455X (gfx1250) — compile-verified, matrix operand order checked by reading, not run on hardware
//
#include <hip/hip_runtime.h>

typedef __attribute__((ext_vector_type(16))) _Float16 v16h;
typedef __attribute__((ext_vector_type(8)))  _Float16 v8h;
typedef __attribute__((ext_vector_type(8)))  float    v8f;
typedef __attribute__((ext_vector_type(4)))  unsigned u32x4;
typedef __attribute__((ext_vector_type(4)))  int      i32x4;
typedef __attribute__((ext_vector_type(8)))  int      i32x8;

#define S_Q   2048
#define NB    2
#define NH    16
#define DD    128
#define BQ    64          // q rows per block (4 waves x 16 rows)
#define BK    32          // key block
#define QK_SCALE 0.08838834764831845f   // 1/sqrt(128), folded into Q

// D = A(16x32 f16) * B(32x16 f16) + C(16x16 f32)
static __device__ __forceinline__ v8f wmma_f16(v16h a, v16h b, v8f c) {
    return __builtin_amdgcn_wmma_f32_16x16x32_f16(
        false, a, false, b, (short)0, c, false, false);
}

// Fragment load from LDS; caller passes p = row_base + lane K-base, 16B aligned.
static __device__ __forceinline__ v16h lds_frag(const _Float16* p) {
    v8h lo = *(const v8h*)(p);
    v8h hi = *(const v8h*)(p + 16);
    return __builtin_shufflevector(lo, hi,
        0, 1, 2, 3, 4, 5, 6, 7, 8, 9, 10, 11, 12, 13, 14, 15);
}

// ---- Tensor Data Mover: async 2D tile load (global f16 -> LDS) ----------
// Builds D# groups per CDNA5 ISA and issues
// tensor_load_to_lds. tile_d0 = contiguous dim (elements), tile_d1 = rows,
// stride0 = row stride (elements). data_size = 2 bytes.
static __device__ __forceinline__ void tdm_load_tile(
        const _Float16* lds_dst, const _Float16* gsrc,
        unsigned tile_d0, unsigned tile_d1, unsigned stride0) {
    unsigned long long ga = (unsigned long long)(uintptr_t)gsrc;
    u32x4 g0;
    g0[0] = 1u;                                        // count=1, user D#
    g0[1] = (unsigned)(uintptr_t)lds_dst;              // LDS byte address
    g0[2] = (unsigned)(ga & 0xFFFFFFFFu);              // global addr [31:0]
    g0[3] = (unsigned)((ga >> 32) & 0x01FFFFFFu)       // global addr [56:32]
          | 0x80000000u;                               // type=2 ("image")
    i32x8 g1;
    g1[0] = (int)(1u << 16);                           // data_size=1 -> 2B
    g1[1] = (int)((tile_d0 & 0xFFFFu) << 16);          // tensor_dim0[15:0]
    g1[2] = (int)(((tile_d0 >> 16) & 0xFFFFu)          // tensor_dim0[31:16]
          | ((tile_d1 & 0xFFFFu) << 16));              // tensor_dim1[15:0]
    g1[3] = (int)(((tile_d1 >> 16) & 0xFFFFu)          // tensor_dim1[31:16]
          | ((tile_d0 & 0xFFFFu) << 16));              // tile_dim0
    g1[4] = (int)(tile_d1 & 0xFFFFu);                  // tile_dim1; tile_dim2=0
    g1[5] = (int)stride0;                              // tensor_dim0_stride[31:0]
    g1[6] = 0;                                         // stride0[47:32]|stride1[15:0]
    g1[7] = 0;
    i32x4 z4 = {0, 0, 0, 0};
#if __has_include(<hip/amd_detail/amd_gfx1250_TDM.h>)
    i32x8 z8 = {0, 0, 0, 0, 0, 0, 0, 0};               // clang-23 6-arg form
    __builtin_amdgcn_tensor_load_to_lds(g0, g1, z4, z4, z8, 0);
#else
    __builtin_amdgcn_tensor_load_to_lds(g0, g1, z4, z4, 0); // ROCm 7.2 5-arg
#endif
}

// ---- pre-pass: K -> f16 [t][b][d]; V -> f16 transposed [b][d][t] ---------
__global__ __launch_bounds__(256)
void cvt_kv_kernel(const float* __restrict__ k, const float* __restrict__ v,
                   _Float16* __restrict__ kh, _Float16* __restrict__ vt) {
    const int i  = blockIdx.x * 256 + threadIdx.x;   // over S_Q*NB*DD
    const int d  = i & (DD - 1);
    const int sb = i >> 7;                           // s*NB + b
    const int bb = sb & (NB - 1);
    const int s  = sb >> 1;
    kh[i] = (_Float16)k[i];
    vt[((size_t)bb * DD + d) * S_Q + s] = (_Float16)v[i];
}

__global__ __launch_bounds__(128)
void fa_sink_mqa_kernel(const float* __restrict__ q,
                        const _Float16* __restrict__ kh,
                        const _Float16* __restrict__ vt,
                        const float* __restrict__ sinks,
                        float* __restrict__ out) {
    __shared__ __align__(16) _Float16 sK [2][BK * DD];   // [key][d] row-major
    __shared__ __align__(16) _Float16 sVT[2][DD * BK];   // [d][key] (V transposed)
    __shared__ __align__(16) _Float16 sP [4][16 * BK];   // per-wave P scratch

    const int tid  = threadIdx.x;
    const int lane = tid & 31;
    const int wave = tid >> 5;
    const int half = lane >> 4;
    const int l16  = lane & 15;
    const int kb8  = half * 8;

    const int b = blockIdx.y;
    const int h = blockIdx.z;
    const int q0    = blockIdx.x * BQ;
    const int qrow0 = q0 + wave * 16;

    // ---- Q stripe -> 4 f16 A-fragments, softmax scale folded in ----
    v16h qf[4];
    {
        const int qr = qrow0 + l16;
        const float* qp = q + (((size_t)qr * NB + b) * NH + h) * DD;
        #pragma unroll
        for (int f = 0; f < 4; ++f) {
            const float* p = qp + f * 32 + kb8;
            #pragma unroll
            for (int e = 0; e < 8; ++e) qf[f][e]     = (_Float16)(p[e]      * QK_SCALE);
            #pragma unroll
            for (int e = 0; e < 8; ++e) qf[f][8 + e] = (_Float16)(p[16 + e] * QK_SCALE);
        }
    }

    const float sink = sinks[h];
    float mrow[8], lrow[8];
    #pragma unroll
    for (int r = 0; r < 8; ++r) { mrow[r] = sink; lrow[r] = 1.0f; }

    v8f o[8];
    v8f zero = {};
    #pragma unroll
    for (int n = 0; n < 8; ++n) o[n] = zero;

    const int nblk = (q0 + BQ) / BK;

    // prologue: TDM-stage block 0 into buffer 0
    if (wave == 0) {
        tdm_load_tile(&sK [0][0], kh + (size_t)(0 * NB + b) * DD, DD, BK, NB * DD);
        tdm_load_tile(&sVT[0][0], vt + (size_t)b * DD * S_Q + 0,  BK, DD, S_Q);
        __builtin_amdgcn_s_wait_tensorcnt(0);
    }
    __syncthreads();

    for (int blk = 0; blk < nblk; ++blk) {
        const int p  = blk & 1;
        const int kb = blk * BK;

        // overlap: DMA next tile into the other buffer while computing this one
        if (wave == 0 && blk + 1 < nblk) {
            const int kn = kb + BK;
            tdm_load_tile(&sK [p ^ 1][0], kh + (size_t)(kn * NB + b) * DD, DD, BK, NB * DD);
            tdm_load_tile(&sVT[p ^ 1][0], vt + (size_t)b * DD * S_Q + kn,  BK, DD, S_Q);
        }

        if (kb <= qrow0 + 15) {   // wave-uniform: skip fully-masked blocks
            // ---- S = Q K^T : preload all B fragments, then WMMA chain ----
            v16h bk[8];
            #pragma unroll
            for (int f = 0; f < 4; ++f) {
                bk[f]     = lds_frag(&sK[p][(l16)      * DD + f * 32 + kb8]);
                bk[4 + f] = lds_frag(&sK[p][(16 + l16) * DD + f * 32 + kb8]);
            }
            v8f c0 = zero, c1 = zero;
            #pragma unroll
            for (int f = 0; f < 4; ++f) c0 = wmma_f16(qf[f], bk[f],     c0);
            #pragma unroll
            for (int f = 0; f < 4; ++f) c1 = wmma_f16(qf[f], bk[4 + f], c1);

            const bool needMask = (kb + 31) > qrow0;  // only diagonal blocks

            #pragma unroll
            for (int r = 0; r < 8; ++r) {
                float x0 = c0[r];
                float x1 = c1[r];
                if (needMask) {
                    const int qr = qrow0 + r + 8 * half;
                    if (kb + l16      > qr) x0 = -3.0e38f;
                    if (kb + 16 + l16 > qr) x1 = -3.0e38f;
                }
                float mx = fmaxf(x0, x1);
                #pragma unroll
                for (int off = 1; off < 16; off <<= 1)
                    mx = fmaxf(mx, __shfl_xor(mx, off, 32));

                const float mnew  = fmaxf(mrow[r], mx);
                const float alpha = __expf(mrow[r] - mnew);
                const float p0 = __expf(x0 - mnew);
                const float p1 = __expf(x1 - mnew);

                float sum = p0 + p1;
                #pragma unroll
                for (int off = 1; off < 16; off <<= 1)
                    sum += __shfl_xor(sum, off, 32);

                lrow[r] = lrow[r] * alpha + sum;
                mrow[r] = mnew;

                #pragma unroll
                for (int n = 0; n < 8; ++n) o[n][r] *= alpha;

                _Float16* pp = &sP[wave][(r + 8 * half) * BK];
                pp[l16]      = (_Float16)p0;
                pp[16 + l16] = (_Float16)p1;
            }

            // ---- O += P V : P as A-fragment (wave-local LDS round trip) ----
            v16h pa = lds_frag(&sP[wave][l16 * BK + kb8]);
            #pragma unroll
            for (int n2 = 0; n2 < 8; ++n2) {
                v16h bv = lds_frag(&sVT[p][(n2 * 16 + l16) * BK + kb8]);
                o[n2] = wmma_f16(pa, bv, o[n2]);
            }
        }

        if (wave == 0) __builtin_amdgcn_s_wait_tensorcnt(0);
        __syncthreads();
    }

    // ---- epilogue ----
    #pragma unroll
    for (int r = 0; r < 8; ++r) {
        const float inv = 1.0f / lrow[r];
        const int qr = qrow0 + r + 8 * half;
        float* op = out + ((size_t)qr * NB + b) * (NH * DD) + h * DD;
        #pragma unroll
        for (int n2 = 0; n2 < 8; ++n2)
            op[n2 * 16 + l16] = o[n2][r] * inv;
    }
}

extern "C" void kernel_launch(void* const* d_in, const int* in_sizes, int n_in,
                              void* d_out, int out_size, void* d_ws, size_t ws_size,
                              hipStream_t stream) {
    const float* q     = (const float*)d_in[0];
    const float* k     = (const float*)d_in[1];
    const float* v     = (const float*)d_in[2];
    const float* sinks = (const float*)d_in[3];
    float* out = (float*)d_out;

    _Float16* kh = (_Float16*)d_ws;                 // [S_Q][NB][DD] f16 (1 MB)
    _Float16* vt = kh + (size_t)S_Q * NB * DD;      // [NB][DD][S_Q] f16 (1 MB)

    cvt_kv_kernel<<<dim3((S_Q * NB * DD) / 256), dim3(256), 0, stream>>>(k, v, kh, vt);

    dim3 grid(S_Q / BQ, NB, NH);   // (32, 2, 16)
    fa_sink_mqa_kernel<<<grid, dim3(128), 0, stream>>>(q, kh, vt, sinks, out);
}
